// DiffeqExactTraceMLP_51178830299530
// MI455X (gfx1250) — matrix operand order repaired and checked by reading, not compiled, not run on hardware
//
#include <hip/hip_runtime.h>

typedef __attribute__((ext_vector_type(16))) __bf16 v16bf;
typedef __attribute__((ext_vector_type(8)))  float  v8f;
typedef __attribute__((ext_vector_type(4)))  unsigned int u32x4;
typedef __attribute__((ext_vector_type(8)))  int i32x8;
typedef __attribute__((ext_vector_type(4)))  int i32x4;

union FragBF { v16bf v; uint4 u[2]; };

__device__ __forceinline__ v8f wmma_bf16(v16bf a, v16bf b, v8f c) {
  // D = A(16x32 bf16) x B(32x16 bf16) + C(16x16 f32)
  return __builtin_amdgcn_wmma_f32_16x16x32_bf16(false, a, false, b, (short)0, c,
                                                 false, false);
}

__device__ __forceinline__ v8f zero8() {
  v8f z;
#pragma unroll
  for (int i = 0; i < 8; ++i) z[i] = 0.f;
  return z;
}

// ---------------------------------------------------------------------------
// Tensor Data Mover: 2-D tile load Global -> LDS (bf16 elements, 2 bytes).
// Builds D# groups per CDNA5 ISA and issues the
// 6-arg (clang-23 / therock) tensor_load_to_lds builtin. Tracked by TENSORcnt.
//   ldsAddr   : LDS byte offset (low 32 bits of generic LDS address)
//   gaddrP    : global byte address of tile start
//   tileX     : tile width  in elements (dim0)
//   tileY     : tile height in rows     (dim1)
//   tensorX/Y : tensor extents for OOB checks (elements / rows)
//   strideX   : row stride in elements (tensor_dim0_stride)
// ---------------------------------------------------------------------------
__device__ __forceinline__ void tdm_load_2d_bf16(unsigned ldsAddr,
                                                 const void* gaddrP,
                                                 unsigned tileX, unsigned tileY,
                                                 unsigned tensorX,
                                                 unsigned tensorY,
                                                 unsigned strideX) {
  unsigned long long ga = (unsigned long long)(uintptr_t)gaddrP;
  u32x4 g0;
  g0[0] = 1u;                                           // count=1 (valid user D#)
  g0[1] = ldsAddr;                                      // lds_addr [63:32]
  g0[2] = (unsigned)(ga & 0xffffffffu);                 // global_addr low
  g0[3] = (unsigned)((ga >> 32) & 0x01ffffffu) | (2u << 30);  // addr hi | type=2
  i32x8 g1;
  g1[0] = (int)(1u << 16);                              // data_size=1 -> 2 bytes
  g1[1] = (int)((tensorX & 0xffffu) << 16);             // tensor_dim0[15:0]@48
  g1[2] = (int)((tensorX >> 16) | ((tensorY & 0xffffu) << 16));  // d0 hi | d1 lo
  g1[3] = (int)((tensorY >> 16) | (tileX << 16));       // d1 hi | tile_dim0
  g1[4] = (int)(tileY & 0xffffu);                       // tile_dim1 (tile_dim2=0)
  g1[5] = (int)strideX;                                 // tensor_dim0_stride lo
  g1[6] = 0;                                            // stride hi | d1stride lo
  g1[7] = 0;
  i32x4 z4;
  z4[0] = z4[1] = z4[2] = z4[3] = 0;
  i32x8 z8;
#pragma unroll
  for (int i = 0; i < 8; ++i) z8[i] = 0;
  __builtin_amdgcn_tensor_load_to_lds(g0, g1, z4, z4, z8, 0);
}

// A-fragment (16x32, 16-bit): lane L holds row (L&15); K chunks of 8 at
// kBase + (L>=16 ? 8 : 0) and +16 further (per CDNA5 ISA A layout).
__device__ __forceinline__ v16bf frag_a(const __bf16* base, int row,
                                        int strideElems, int kBase, int lane) {
  const __bf16* p = base + row * strideElems + kBase + ((lane & 16) >> 1);
  FragBF f;
  f.u[0] = *(const uint4*)(p);
  f.u[1] = *(const uint4*)(p + 16);
  return f.v;
}

// B-fragment (32x16, 16-bit) read from N-major (transposed) storage:
// lane L holds column (L&15); K chunks of 8 at kBase + (L>=16 ? 8 : 0) and +16 further.
__device__ __forceinline__ v16bf frag_b(const __bf16* base, int row,
                                        int strideElems, int kBase, int lane) {
  const __bf16* p = base + row * strideElems + kBase + ((lane & 16) >> 1);
  FragBF f;
  f.u[0] = *(const uint4*)(p);
  f.u[1] = *(const uint4*)(p + 16);
  return f.v;
}

// ---------------------------------------------------------------------------
// Generic bf16 WMMA GEMM: out = epi(A[MxK] @ BT[NxK]^T + bias)
// EPI 0: relu, EPI 1: bias only, EPI 2: + addin (element-wise), all -> bf16
// Workgroup tile 128x128, 8 waves (2x4), each wave 64x32, K-step 32.
// ---------------------------------------------------------------------------
template <int EPI>
__global__ __launch_bounds__(256) void gemm_bf16_wmma(
    const __bf16* __restrict__ A, const __bf16* __restrict__ BT,
    const float* __restrict__ bias, const __bf16* __restrict__ addin,
    __bf16* __restrict__ out, int M, int N, int K) {
  __shared__ __bf16 sA[2][128 * 32];
  __shared__ __bf16 sB[2][128 * 32];
  const int tid = threadIdx.x;
  const int lane = tid & 31, wid = tid >> 5;
  const int waveM = wid & 1, waveN = wid >> 1;
  const int bM = blockIdx.y * 128, bN = blockIdx.x * 128;
  const int colAdd = lane & 15, rowAdd = (lane >> 4) << 3;

  v8f acc[4][2];
#pragma unroll
  for (int mt = 0; mt < 4; ++mt)
#pragma unroll
    for (int nt = 0; nt < 2; ++nt) acc[mt][nt] = zero8();

  const int nK = K >> 5;
  auto stage = [&](int kt, int b) {
    const int k0 = kt * 32;
#pragma unroll
    for (int i = 0; i < 2; ++i) {
      int c = tid + 256 * i;  // 512 16-byte chunks per operand tile
      int row = c >> 2, kc = (c & 3) << 3;
      const __bf16* pa = A + (size_t)(bM + row) * K + k0 + kc;
      const __bf16* pb = BT + (size_t)(bN + row) * K + k0 + kc;
      *(uint4*)(&sA[b][row * 32 + kc]) = *(const uint4*)pa;
      *(uint4*)(&sB[b][row * 32 + kc]) = *(const uint4*)pb;
      if (k0 + 32 < K) {  // keep HBM/L2 ahead of the double buffer
        __builtin_prefetch((const void*)(pa + 32), 0, 0);
        __builtin_prefetch((const void*)(pb + 32), 0, 0);
      }
    }
  };
  stage(0, 0);
  __syncthreads();
  for (int kt = 0; kt < nK; ++kt) {
    const int cb = kt & 1;
    if (kt + 1 < nK) stage(kt + 1, cb ^ 1);
    v16bf af[4], bf2[2];
#pragma unroll
    for (int mt = 0; mt < 4; ++mt)
      af[mt] = frag_a(sA[cb], waveM * 64 + mt * 16 + colAdd, 32, 0, lane);
#pragma unroll
    for (int nt = 0; nt < 2; ++nt)
      bf2[nt] = frag_b(sB[cb], waveN * 32 + nt * 16 + colAdd, 32, 0, lane);
#pragma unroll
    for (int mt = 0; mt < 4; ++mt)
#pragma unroll
      for (int nt = 0; nt < 2; ++nt)
        acc[mt][nt] = wmma_bf16(af[mt], bf2[nt], acc[mt][nt]);
    __syncthreads();
  }
  // Epilogue: C layout -> VGPR g: lanes 0-15 row g, lanes 16-31 row g+8; N=lane%16
#pragma unroll
  for (int mt = 0; mt < 4; ++mt) {
#pragma unroll
    for (int nt = 0; nt < 2; ++nt) {
      const int gn = bN + waveN * 32 + nt * 16 + colAdd;
      const float bv = bias ? bias[gn] : 0.f;
#pragma unroll
      for (int g = 0; g < 8; ++g) {
        const int gm = bM + waveM * 64 + mt * 16 + rowAdd + g;
        float v = acc[mt][nt][g] + bv;
        if (EPI == 0) v = fmaxf(v, 0.f);
        if (EPI == 2) v += (float)addin[(size_t)gm * N + gn];
        out[(size_t)gm * N + gn] = (__bf16)v;
      }
    }
  }
}

// ---------------------------------------------------------------------------
// Prep kernels: MADE masks (natural/reversed degrees), bf16 convert, transpose
// ---------------------------------------------------------------------------
__global__ void k_mask_w0(const float* __restrict__ W, __bf16* __restrict__ WT,
                          int rev) {
  int idx = blockIdx.x * 256 + threadIdx.x;  // 512*128
  if (idx >= 512 * 128) return;
  int j = idx >> 7, d = idx & 127;
  int degin = rev ? (127 - d) : d;
  float v = (degin <= (j % 127)) ? W[d * 512 + j] : 0.f;
  WT[j * 128 + d] = (__bf16)v;
}

__global__ void k_mask_w1(const float* __restrict__ W, __bf16* __restrict__ WT) {
  int idx = blockIdx.x * 256 + threadIdx.x;  // 512*512
  if (idx >= 512 * 512) return;
  int j = idx >> 9, i = idx & 511;
  float v = ((i % 127) <= (j % 127)) ? W[i * 512 + j] : 0.f;
  WT[j * 512 + i] = (__bf16)v;
}

__global__ void k_mask_w2(const float* __restrict__ W, __bf16* __restrict__ WT,
                          int rev) {
  int idx = blockIdx.x * 256 + threadIdx.x;  // 8192*512
  if (idx >= 8192 * 512) return;
  int c = idx >> 9, i = idx & 511;
  int d = c & 127;
  int degin = rev ? (127 - d) : d;
  float v = ((i % 127) < degin) ? W[(size_t)i * 8192 + c] : 0.f;
  WT[(size_t)c * 512 + i] = (__bf16)v;
}

__global__ void k_f32_to_bf16(const float* __restrict__ src,
                              __bf16* __restrict__ dst, int n) {
  int idx = blockIdx.x * 256 + threadIdx.x;
  if (idx < n) dst[idx] = (__bf16)src[idx];
}

__global__ void k_prep_dw0h(const float* __restrict__ dW0,
                            __bf16* __restrict__ WT) {
  int idx = blockIdx.x * 256 + threadIdx.x;  // 512*64 ; WT[j][k] = dW0[k+2][j]
  if (idx >= 512 * 64) return;
  int j = idx >> 6, k = idx & 63;
  WT[j * 64 + k] = (__bf16)dW0[(k + 2) * 512 + j];
}

__global__ void k_transpose_bf16(const float* __restrict__ W,
                                 __bf16* __restrict__ WT) {
  int idx = blockIdx.x * 256 + threadIdx.x;  // 512*512 ; WT[j][i] = W[i][j]
  if (idx >= 512 * 512) return;
  int j = idx >> 9, i = idx & 511;
  WT[j * 512 + i] = (__bf16)W[i * 512 + j];
}

// Hg[(b*128+d)*64 + dh] = Hsum[b][dh*128 + d]
__global__ void k_gather_h(const __bf16* __restrict__ Hs,
                           __bf16* __restrict__ Hg) {
  int idx = blockIdx.x * 256 + threadIdx.x;  // 1024*128*64
  if (idx >= 1024 * 128 * 64) return;
  int b = idx >> 13, rem = idx & 8191, d = rem >> 6, dh = rem & 63;
  Hg[idx] = Hs[(size_t)b * 8192 + dh * 128 + d];
}

// ---------------------------------------------------------------------------
// Fused dimwise MLP + exact JVP. 64 rows per workgroup, 8 waves.
// Phase A: a0 = Hg(64x64)@W0h(64x512) + t*W0[0] + x*W0[1] + b0 -> z0,tz0 (LDS)
// Phase B: per 128-col chunk: a1 = z0@W1, t1 = tz0@W1 (WMMA pairs), then
//          z1 = tanh(a1+b1), tz1 = t1*(1-z1^2), dot with W2 -> y, jac.
// All Global->LDS staging in this kernel goes through the Tensor Data Mover
// (one issue per workgroup by wave 0; completion via s_wait_tensorcnt).
// ---------------------------------------------------------------------------
__global__ __launch_bounds__(256) void dimwise_jvp(
    const __bf16* __restrict__ Hg, const float* __restrict__ x,
    const float* __restrict__ t, const float* __restrict__ dW0,
    const float* __restrict__ b0, const __bf16* __restrict__ W0hT,
    const __bf16* __restrict__ W1T, const float* __restrict__ b1,
    const float* __restrict__ W2, const float* __restrict__ b2,
    float* __restrict__ y, float* __restrict__ jac) {
  extern __shared__ char smem[];
  __bf16* Z0 = (__bf16*)smem;      // [64][512] z0 (bf16)
  __bf16* T0 = Z0 + 64 * 512;      // [64][512] tangent of z0
  __bf16* WB = T0 + 64 * 512;      // phase A: W0hT [512][64]; B: 2x[128][32]
  __bf16* AT = WB + 512 * 64;      // [64][64] Hg tile
  float* yL = (float*)(AT + 64 * 64);  // [64]
  float* jL = yL + 64;                 // [64]

  const int tid = threadIdx.x, lane = tid & 31, wid = tid >> 5;
  const int r0 = blockIdx.x * 64;
  const int colAdd = lane & 15, rowAdd = (lane >> 4) << 3;

  if (tid < 64) { yL[tid] = 0.f; jL[tid] = 0.f; }
  // TDM staging: Hg tile (64x64) and the full W0hT block (512x64).
  if (wid == 0) {
    tdm_load_2d_bf16((unsigned)(uintptr_t)AT, Hg + (size_t)r0 * 64,
                     /*tileX=*/64, /*tileY=*/64, /*tensorX=*/64,
                     /*tensorY=*/1u << 20, /*strideX=*/64);
    tdm_load_2d_bf16((unsigned)(uintptr_t)WB, W0hT,
                     /*tileX=*/64, /*tileY=*/512, /*tensorX=*/64,
                     /*tensorY=*/512, /*strideX=*/64);
    __builtin_amdgcn_s_wait_tensorcnt(0);
  }
  __syncthreads();

  // ---- Phase A: each wave owns 64 output columns (wid*64..), all 64 rows ----
  v8f acc0[4][4];
#pragma unroll
  for (int mt = 0; mt < 4; ++mt)
#pragma unroll
    for (int nt = 0; nt < 4; ++nt) acc0[mt][nt] = zero8();
#pragma unroll
  for (int ks = 0; ks < 2; ++ks) {
    const int kb = ks * 32;
    v16bf af[4];
#pragma unroll
    for (int mt = 0; mt < 4; ++mt)
      af[mt] = frag_a(AT, mt * 16 + colAdd, 64, kb, lane);
#pragma unroll
    for (int nt = 0; nt < 4; ++nt) {
      v16bf bfg = frag_b(WB, wid * 64 + nt * 16 + colAdd, 64, kb, lane);
#pragma unroll
      for (int mt = 0; mt < 4; ++mt)
        acc0[mt][nt] = wmma_bf16(af[mt], bfg, acc0[mt][nt]);
    }
  }
  const float tv = t[0];
#pragma unroll
  for (int mt = 0; mt < 4; ++mt) {
#pragma unroll
    for (int nt = 0; nt < 4; ++nt) {
      const int n = wid * 64 + nt * 16 + colAdd;
      const float w0r0 = dW0[n], w0r1 = dW0[512 + n], bb = b0[n];
#pragma unroll
      for (int g = 0; g < 8; ++g) {
        const int rloc = mt * 16 + rowAdd + g;
        const float a = acc0[mt][nt][g] + tv * w0r0 + x[r0 + rloc] * w0r1 + bb;
        const float z = tanhf(a);
        Z0[rloc * 512 + n] = (__bf16)z;
        T0[rloc * 512 + n] = (__bf16)(w0r1 * (1.f - z * z));
      }
    }
  }
  __syncthreads();

  // ---- Phase B: 4 chunks of 128 cols; each wave owns a 16-col subtile ----
  const float b2v = b2[0];
  auto stageW1 = [&](int nbase, int kt, int b) {
    if (wid == 0) {
      tdm_load_2d_bf16((unsigned)(uintptr_t)(WB + b * (128 * 32)),
                       W1T + (size_t)nbase * 512 + kt * 32,
                       /*tileX=*/32, /*tileY=*/128, /*tensorX=*/512,
                       /*tensorY=*/512, /*strideX=*/512);
    }
  };
  for (int nc = 0; nc < 4; ++nc) {
    const int nbase = nc * 128;
    v8f a1[4], t1[4];
#pragma unroll
    for (int mt = 0; mt < 4; ++mt) { a1[mt] = zero8(); t1[mt] = zero8(); }
    stageW1(nbase, 0, 0);
    if (wid == 0) __builtin_amdgcn_s_wait_tensorcnt(0);
    __syncthreads();
    for (int kt = 0; kt < 16; ++kt) {
      const int cb = kt & 1;
      if (kt + 1 < 16) stageW1(nbase, kt + 1, cb ^ 1);  // TDM overlaps compute
      const int kb = kt * 32;
      const __bf16* cur = WB + cb * (128 * 32);
      v16bf bfg = frag_b(cur, wid * 16 + colAdd, 32, 0, lane);
#pragma unroll
      for (int mt = 0; mt < 4; ++mt) {
        v16bf za = frag_a(Z0, mt * 16 + colAdd, 512, kb, lane);
        v16bf ta = frag_a(T0, mt * 16 + colAdd, 512, kb, lane);
        a1[mt] = wmma_bf16(za, bfg, a1[mt]);
        t1[mt] = wmma_bf16(ta, bfg, t1[mt]);
      }
      if (wid == 0) __builtin_amdgcn_s_wait_tensorcnt(0);
      __syncthreads();
    }
    const int n = nbase + wid * 16 + colAdd;
    const float b1v = b1[n], w2v = W2[n];
#pragma unroll
    for (int mt = 0; mt < 4; ++mt) {
#pragma unroll
      for (int g = 0; g < 8; ++g) {
        const float z = tanhf(a1[mt][g] + b1v);
        const float tz = t1[mt][g] * (1.f - z * z);
        float ys = z * w2v, js = tz * w2v;
#pragma unroll
        for (int off = 1; off < 16; off <<= 1) {
          ys += __shfl_xor(ys, off, 32);
          js += __shfl_xor(js, off, 32);
        }
        if (colAdd == 0) {
          const int rloc = mt * 16 + rowAdd + g;
          atomicAdd(&yL[rloc], ys);   // ds_add_f32
          atomicAdd(&jL[rloc], js);
        }
      }
    }
    __syncthreads();
  }
  if (tid < 64) {
    y[r0 + tid] = yL[tid] + b2v;
    jac[r0 + tid] = jL[tid];
  }
}

// ---------------------------------------------------------------------------
extern "C" void kernel_launch(void* const* d_in, const int* in_sizes, int n_in,
                              void* d_out, int out_size, void* d_ws,
                              size_t ws_size, hipStream_t stream) {
  (void)in_sizes; (void)n_in; (void)out_size; (void)ws_size;
  const float* t = (const float*)d_in[0];
  const float* x = (const float*)d_in[1];
  const float* mW0[2] = {(const float*)d_in[2], (const float*)d_in[8]};
  const float* mb0[2] = {(const float*)d_in[3], (const float*)d_in[9]};
  const float* mW1[2] = {(const float*)d_in[4], (const float*)d_in[10]};
  const float* mb1[2] = {(const float*)d_in[5], (const float*)d_in[11]};
  const float* mW2[2] = {(const float*)d_in[6], (const float*)d_in[12]};
  const float* mb2[2] = {(const float*)d_in[7], (const float*)d_in[13]};
  const float* dW0 = (const float*)d_in[14];
  const float* db0 = (const float*)d_in[15];
  const float* dW1 = (const float*)d_in[16];
  const float* db1 = (const float*)d_in[17];
  const float* dW2 = (const float*)d_in[18];
  const float* db2 = (const float*)d_in[19];

  char* ws = (char*)d_ws;
  size_t off = 0;
  auto alloc = [&](size_t bytes) {
    void* p = ws + off;
    off = (off + bytes + 255) & ~(size_t)255;
    return p;
  };
  __bf16* Xb = (__bf16*)alloc((size_t)1024 * 128 * 2);
  __bf16 *W0T[2], *W1T[2], *W2T[2];
  for (int p = 0; p < 2; ++p) {
    W0T[p] = (__bf16*)alloc((size_t)512 * 128 * 2);
    W1T[p] = (__bf16*)alloc((size_t)512 * 512 * 2);
    W2T[p] = (__bf16*)alloc((size_t)8192 * 512 * 2);
  }
  __bf16* dW0hT = (__bf16*)alloc((size_t)512 * 64 * 2);
  __bf16* dW1T = (__bf16*)alloc((size_t)512 * 512 * 2);
  __bf16* H0 = (__bf16*)alloc((size_t)1024 * 512 * 2);
  __bf16* H1 = (__bf16*)alloc((size_t)1024 * 512 * 2);
  __bf16* Hs = (__bf16*)alloc((size_t)1024 * 8192 * 2);
  __bf16* Hg = (__bf16*)alloc((size_t)131072 * 64 * 2);

  dim3 B256(256);
  k_f32_to_bf16<<<dim3((1024 * 128) / 256), B256, 0, stream>>>(x, Xb,
                                                               1024 * 128);
  for (int p = 0; p < 2; ++p) {
    k_mask_w0<<<dim3((512 * 128) / 256), B256, 0, stream>>>(mW0[p], W0T[p], p);
    k_mask_w1<<<dim3((512 * 512) / 256), B256, 0, stream>>>(mW1[p], W1T[p]);
    k_mask_w2<<<dim3((8192 * 512) / 256), B256, 0, stream>>>(mW2[p], W2T[p], p);
  }
  k_prep_dw0h<<<dim3((512 * 64) / 256), B256, 0, stream>>>(dW0, dW0hT);
  k_transpose_bf16<<<dim3((512 * 512) / 256), B256, 0, stream>>>(dW1, dW1T);

  // Two MADE networks; second L2 accumulates onto the first's output in place.
  for (int p = 0; p < 2; ++p) {
    gemm_bf16_wmma<0><<<dim3(512 / 128, 1024 / 128), B256, 0, stream>>>(
        Xb, W0T[p], mb0[p], nullptr, H0, 1024, 512, 128);
    gemm_bf16_wmma<0><<<dim3(512 / 128, 1024 / 128), B256, 0, stream>>>(
        H0, W1T[p], mb1[p], nullptr, H1, 1024, 512, 512);
    if (p == 0)
      gemm_bf16_wmma<1><<<dim3(8192 / 128, 1024 / 128), B256, 0, stream>>>(
          H1, W2T[p], mb2[p], nullptr, Hs, 1024, 8192, 512);
    else
      gemm_bf16_wmma<2><<<dim3(8192 / 128, 1024 / 128), B256, 0, stream>>>(
          H1, W2T[p], mb2[p], Hs, Hs, 1024, 8192, 512);
  }
  k_gather_h<<<dim3((1024 * 128 * 64) / 256), B256, 0, stream>>>(Hs, Hg);

  float* y = (float*)d_out;
  float* jc = y + 1024 * 128;
  size_t smem = (size_t)(64 * 512 * 2) * 2 + (size_t)512 * 64 * 2 +
                (size_t)64 * 64 * 2 + 2 * 64 * sizeof(float);  // ~200.5 KB
  dimwise_jvp<<<dim3(131072 / 64), B256, smem, stream>>>(
      Hg, x, t, dW0, db0, dW0hT, dW1T, db1, dW2, db2, y, jc);
}
